// RealPhaseRouter_35794257445259
// MI455X (gfx1250) — matrix operand order repaired and checked by reading, compiled, not run on hardware
//
#include <hip/hip_runtime.h>
#include <hip/hip_bf16.h>

typedef __bf16 bf16_t;
typedef __attribute__((ext_vector_type(16))) __bf16 v16bf;
typedef __attribute__((ext_vector_type(8)))  float  v8f;

#define B_      2048   // batch
#define H_      256    // hidden
#define IN_     320    // input dim
#define TH_     768    // 3*H
#define NOPS_   1883
#define NP_     2048   // NOPS padded to 128 tiles of 16
#define OPE_PAD 1920   // op_embeds rows padded to 60 blocks of 32
#define STEPS_  20

// ---------------------------------------------------------------- helpers
__device__ __forceinline__ bf16_t f2bf(float f) {
  unsigned u = __float_as_uint(f);
  unsigned r = (u + 0x7FFFu + ((u >> 16) & 1u)) >> 16;  // RNE truncate
  unsigned short s = (unsigned short)r;
  bf16_t b;
  __builtin_memcpy(&b, &s, 2);
  return b;
}

// ---------------------------------------------------------------- GEMM
// C(M x N) = A(M x K, bf16, row-major) * W(N x K, bf16, row-major)^T + bias
// One wave computes a 32(M) x 64(N) tile (2 M-tiles share each B fragment);
// 4 waves/block -> 32 x 256. grid = (Ntiles/16, M/32), block = 128.
// W rows are padded so fragment loads never go OOB; stores masked by Nvalid.
__global__ __launch_bounds__(128) void gemm_bf16_kernel(
    const bf16_t* __restrict__ A, int lda,
    const bf16_t* __restrict__ W, int ldw,
    const float* __restrict__ bias,
    float* __restrict__ Cf, bf16_t* __restrict__ Cb, int ldc,
    int K, int Nvalid)
{
  const int lane = threadIdx.x & 31;
  const int wv   = threadIdx.x >> 5;
  const int mblk = blockIdx.y;
  const int nt0  = blockIdx.x * 16 + wv * 4;
  const int lo   = lane & 15;
  const int hi   = lane >> 4;

  // A fragment: lane holds row; K pattern [k+8*hi..+7] then [k+16+8*hi..+7]
  const bf16_t* Arow0 = A + (size_t)(mblk * 32 + lo) * lda + (hi << 3);
  const bf16_t* Arow1 = Arow0 + (size_t)16 * lda;
  const int bsh = hi << 3;   // B fragment: K pattern [k+8*hi..+7] then [k+16+8*hi..+7]

  v8f acc[2][4] = {};
  for (int k = 0; k < K; k += 32) {
    union { v16bf v; uint4 u[2]; } fa0, fa1;
    fa0.u[0] = *(const uint4*)(Arow0 + k);
    fa0.u[1] = *(const uint4*)(Arow0 + k + 16);
    fa1.u[0] = *(const uint4*)(Arow1 + k);
    fa1.u[1] = *(const uint4*)(Arow1 + k + 16);
#pragma unroll
    for (int t = 0; t < 4; ++t) {
      const bf16_t* Brow = W + (size_t)((nt0 + t) * 16 + lo) * ldw + k + bsh;
      union { v16bf v; uint4 u[2]; } fb;
      fb.u[0] = *(const uint4*)(Brow);
      fb.u[1] = *(const uint4*)(Brow + 16);
      acc[0][t] = __builtin_amdgcn_wmma_f32_16x16x32_bf16(
          false, fa0.v, false, fb.v, (short)0, acc[0][t], false, false);
      acc[1][t] = __builtin_amdgcn_wmma_f32_16x16x32_bf16(
          false, fa1.v, false, fb.v, (short)0, acc[1][t], false, false);
    }
  }
#pragma unroll
  for (int mi = 0; mi < 2; ++mi) {
    const int mbase = mblk * 32 + mi * 16 + (hi << 3);
#pragma unroll
    for (int t = 0; t < 4; ++t) {
      int n = (nt0 + t) * 16 + lo;
      if (n >= Nvalid) continue;
      float bv = bias ? bias[n] : 0.0f;
#pragma unroll
      for (int v = 0; v < 8; ++v) {
        float val = acc[mi][t][v] + bv;
        size_t idx = (size_t)(mbase + v) * ldc + n;
        if (Cf) Cf[idx] = val; else Cb[idx] = f2bf(val);
      }
    }
  }
}

// ---------------------------------------------------------------- scores
// Dual GEMM sharing A: logits = h @ We^T + be ; p = h @ opp^T.
// scores = logits - 0.5*sqrt(max(|h|^2 + |opp|^2 - 2p, 0)), written to d_out slab.
// One wave: 32(M) x 64(N), both products -> 16 accumulator tiles.
__global__ __launch_bounds__(128) void scores_kernel(
    const bf16_t* __restrict__ Hb,
    const bf16_t* __restrict__ We,    // (NP_, H_) bf16, zero-padded
    const bf16_t* __restrict__ Opp,   // (NP_, H_) bf16, zero-padded
    const float* __restrict__ be,     // (NP_)
    const float* __restrict__ oppsq,  // (NP_)
    const float* __restrict__ hsq,    // (B_)
    float* __restrict__ out)          // (B_, NOPS_) this step
{
  const int lane = threadIdx.x & 31;
  const int wv   = threadIdx.x >> 5;
  const int mblk = blockIdx.y;
  const int nt0  = blockIdx.x * 16 + wv * 4;
  const int lo   = lane & 15;
  const int hi   = lane >> 4;
  const bf16_t* Arow0 = Hb + (size_t)(mblk * 32 + lo) * H_ + (hi << 3);
  const bf16_t* Arow1 = Arow0 + (size_t)16 * H_;
  const int bsh = hi << 3;

  v8f accL[2][4] = {};
  v8f accP[2][4] = {};
  for (int k = 0; k < H_; k += 32) {
    union { v16bf v; uint4 u[2]; } fa0, fa1;
    fa0.u[0] = *(const uint4*)(Arow0 + k);
    fa0.u[1] = *(const uint4*)(Arow0 + k + 16);
    fa1.u[0] = *(const uint4*)(Arow1 + k);
    fa1.u[1] = *(const uint4*)(Arow1 + k + 16);
#pragma unroll
    for (int t = 0; t < 4; ++t) {
      size_t wr = (size_t)((nt0 + t) * 16 + lo) * H_ + k + bsh;
      union { v16bf v; uint4 u[2]; } fb;
      fb.u[0] = *(const uint4*)(We + wr);
      fb.u[1] = *(const uint4*)(We + wr + 16);
      accL[0][t] = __builtin_amdgcn_wmma_f32_16x16x32_bf16(
          false, fa0.v, false, fb.v, (short)0, accL[0][t], false, false);
      accL[1][t] = __builtin_amdgcn_wmma_f32_16x16x32_bf16(
          false, fa1.v, false, fb.v, (short)0, accL[1][t], false, false);
      fb.u[0] = *(const uint4*)(Opp + wr);
      fb.u[1] = *(const uint4*)(Opp + wr + 16);
      accP[0][t] = __builtin_amdgcn_wmma_f32_16x16x32_bf16(
          false, fa0.v, false, fb.v, (short)0, accP[0][t], false, false);
      accP[1][t] = __builtin_amdgcn_wmma_f32_16x16x32_bf16(
          false, fa1.v, false, fb.v, (short)0, accP[1][t], false, false);
    }
  }
#pragma unroll
  for (int mi = 0; mi < 2; ++mi) {
    const int mbase = mblk * 32 + mi * 16 + (hi << 3);
#pragma unroll
    for (int t = 0; t < 4; ++t) {
      int n = (nt0 + t) * 16 + lo;
      if (n >= NOPS_) continue;
      float bn = be[n], oq = oppsq[n];
#pragma unroll
      for (int v = 0; v < 8; ++v) {
        int m = mbase + v;
        float d2 = hsq[m] + oq - 2.0f * accP[mi][t][v];
        d2 = fmaxf(d2, 0.0f);
        out[(size_t)m * NOPS_ + n] = accL[mi][t][v] + bn - 0.5f * __builtin_sqrtf(d2);
      }
    }
  }
}

// ---------------------------------------------------------------- GRU gates
// One block per row. GI/GH hold x@Wih^T+bih and h@Whh^T+bhh. In-place h update.
__global__ __launch_bounds__(256) void gates_kernel(
    const float* __restrict__ GI, const float* __restrict__ GH,
    float* __restrict__ h, bf16_t* __restrict__ hb,
    float* __restrict__ hsq)
{
  __shared__ float red[256];
  const int m = blockIdx.x, c = threadIdx.x;
  const size_t gb = (size_t)m * TH_;
  float gir = GI[gb + c], giz = GI[gb + H_ + c], gin = GI[gb + 2 * H_ + c];
  float ghr = GH[gb + c], ghz = GH[gb + H_ + c], ghn = GH[gb + 2 * H_ + c];
  float r  = 1.0f / (1.0f + __expf(-(gir + ghr)));
  float z  = 1.0f / (1.0f + __expf(-(giz + ghz)));
  float nn = tanhf(gin + r * ghn);
  const size_t hi = (size_t)m * H_ + c;
  float hp = h[hi];
  float hv = (1.0f - z) * nn + z * hp;
  h[hi]  = hv;
  hb[hi] = f2bf(hv);
  if (hsq) {
    red[c] = hv * hv;
    __syncthreads();
    for (int s = 128; s > 0; s >>= 1) {
      if (c < s) red[c] += red[c + s];
      __syncthreads();
    }
    if (c == 0) hsq[m] = red[0];
  }
}

// ---------------------------------------------------------------- opp stats
__global__ __launch_bounds__(256) void oppstat_kernel(
    const float* __restrict__ oppf, bf16_t* __restrict__ oppb,
    float* __restrict__ opp_sq)
{
  __shared__ float red[256];
  const int m = blockIdx.x, c = threadIdx.x;
  const size_t i = (size_t)m * H_ + c;
  float v = oppf[i];
  oppb[i] = f2bf(v);
  red[c] = v * v;
  __syncthreads();
  for (int s = 128; s > 0; s >>= 1) {
    if (c < s) red[c] += red[c + s];
    __syncthreads();
  }
  if (c == 0) opp_sq[m] = red[0];
}

// ---------------------------------------------------------------- argmax + gather
// One wave per row: strided scan + shuffle butterfly with first-index tiebreak,
// then gather op_embeds[sel] and build bf16 [h, sel_emb] concat for proj GEMM.
__global__ __launch_bounds__(256) void argmax_kernel(
    const float* __restrict__ scores,  // (B_, NOPS_) this step
    int* __restrict__ traj,            // (B_) this step
    const bf16_t* __restrict__ h1b,
    const float* __restrict__ op_embeds,
    bf16_t* __restrict__ xcat)
{
  const int lane = threadIdx.x & 31;
  const int wv   = threadIdx.x >> 5;
  const int m    = blockIdx.x * 8 + wv;
  const float* row = scores + (size_t)m * NOPS_;
  float bestv = -__builtin_inff();
  int besti = 0;
  for (int i = lane; i < NOPS_; i += 32) {
    float v = row[i];
    if (v > bestv) { bestv = v; besti = i; }
  }
  for (int off = 16; off > 0; off >>= 1) {
    float ov = __shfl_xor(bestv, off, 32);
    int   oi = __shfl_xor(besti, off, 32);
    if (ov > bestv || (ov == bestv && oi < besti)) { bestv = ov; besti = oi; }
  }
  if (lane == 0) traj[m] = besti;
  const float*  emb = op_embeds + (size_t)besti * 256;
  bf16_t*       xr  = xcat + (size_t)m * 512;
  const bf16_t* hr  = h1b + (size_t)m * H_;
  for (int j = lane; j < 256; j += 32) {
    xr[j]       = hr[j];
    xr[256 + j] = f2bf(emb[j]);
  }
}

// ---------------------------------------------------------------- small prep kernels
__global__ void cvt_pad_kernel(bf16_t* dst, const float* src, int R, int C, int total)
{
  int i = blockIdx.x * blockDim.x + threadIdx.x;
  if (i >= total) return;
  int r = i / C;
  dst[i] = (r < R) ? f2bf(src[i]) : f2bf(0.0f);
}

__global__ void pad_bias_kernel(float* dst, const float* src, int N, int Npad)
{
  int i = blockIdx.x * blockDim.x + threadIdx.x;
  if (i >= Npad) return;
  dst[i] = (i < N) ? src[i] : 0.0f;
}

__global__ void zero_kernel(unsigned* p, int n)
{
  int i = blockIdx.x * blockDim.x + threadIdx.x;
  if (i < n) p[i] = 0u;
}

__global__ void concat_x0_kernel(bf16_t* xb, const float* geo, const float* sem)
{
  int i = blockIdx.x * blockDim.x + threadIdx.x;
  if (i >= B_ * IN_) return;
  int m = i / IN_, c = i % IN_;
  float v = (c < 256) ? geo[(size_t)m * 256 + c] : sem[(size_t)m * 64 + (c - 256)];
  xb[i] = f2bf(v);
}

// ---------------------------------------------------------------- launch
extern "C" void kernel_launch(void* const* d_in, const int* in_sizes, int n_in,
                              void* d_out, int out_size, void* d_ws, size_t ws_size,
                              hipStream_t stream) {
  const float* geo      = (const float*)d_in[0];
  const float* sem      = (const float*)d_in[1];
  const float* w_ih0    = (const float*)d_in[2];
  const float* w_hh0    = (const float*)d_in[3];
  const float* b_ih0    = (const float*)d_in[4];
  const float* b_hh0    = (const float*)d_in[5];
  const float* w_ih1    = (const float*)d_in[6];
  const float* w_hh1    = (const float*)d_in[7];
  const float* b_ih1    = (const float*)d_in[8];
  const float* b_hh1    = (const float*)d_in[9];
  const float* w_energy = (const float*)d_in[10];
  const float* b_energy = (const float*)d_in[11];
  const float* w_proj   = (const float*)d_in[12];
  const float* b_proj   = (const float*)d_in[13];
  const float* w_op     = (const float*)d_in[14];
  const float* op_emb   = (const float*)d_in[15];

  char* ws = (char*)d_ws;
  size_t off = 0;
  auto alloc = [&](size_t bytes) { size_t o = off; off += (bytes + 255) & ~(size_t)255; return o; };

  // bf16 buffers
  bf16_t* wih0b  = (bf16_t*)(ws + alloc((size_t)TH_ * IN_ * 2));
  bf16_t* whh0b  = (bf16_t*)(ws + alloc((size_t)TH_ * H_ * 2));
  bf16_t* wih1b  = (bf16_t*)(ws + alloc((size_t)TH_ * H_ * 2));
  bf16_t* whh1b  = (bf16_t*)(ws + alloc((size_t)TH_ * H_ * 2));
  bf16_t* web    = (bf16_t*)(ws + alloc((size_t)NP_ * H_ * 2));
  bf16_t* wprojb = (bf16_t*)(ws + alloc((size_t)512 * 512 * 2));
  bf16_t* opeb   = (bf16_t*)(ws + alloc((size_t)OPE_PAD * 256 * 2));
  bf16_t* wopb   = (bf16_t*)(ws + alloc((size_t)H_ * 256 * 2));
  bf16_t* oppb   = (bf16_t*)(ws + alloc((size_t)NP_ * H_ * 2));
  bf16_t* xb     = (bf16_t*)(ws + alloc((size_t)B_ * IN_ * 2));
  bf16_t* h0b    = (bf16_t*)(ws + alloc((size_t)B_ * H_ * 2));
  bf16_t* h1b    = (bf16_t*)(ws + alloc((size_t)B_ * H_ * 2));
  bf16_t* xcatb  = (bf16_t*)(ws + alloc((size_t)B_ * 512 * 2));
  // f32 buffers
  float* GI     = (float*)(ws + alloc((size_t)B_ * TH_ * 4));
  float* GH     = (float*)(ws + alloc((size_t)B_ * TH_ * 4));
  float* h0f    = (float*)(ws + alloc((size_t)B_ * H_ * 4));
  float* h1f    = (float*)(ws + alloc((size_t)B_ * H_ * 4));
  float* hsq    = (float*)(ws + alloc((size_t)B_ * 4));
  float* oppf   = (float*)(ws + alloc((size_t)NP_ * H_ * 4));
  float* oppsq  = (float*)(ws + alloc((size_t)NP_ * 4));
  float* bep    = (float*)(ws + alloc((size_t)NP_ * 4));
  float* bprojp = (float*)(ws + alloc((size_t)512 * 4));

  float* out = (float*)d_out;
  int*   traj_base   = (int*)d_out;                 // (STEPS_, B_) int32
  float* scores_base = out + (size_t)STEPS_ * B_;   // (STEPS_, B_, NOPS_)

  auto ew = [&](int n) { return dim3((unsigned)((n + 255) / 256)); };

  // ---- prep: weight conversion / padding
  cvt_pad_kernel<<<ew(TH_ * IN_), 256, 0, stream>>>(wih0b, w_ih0, TH_, IN_, TH_ * IN_);
  cvt_pad_kernel<<<ew(TH_ * H_), 256, 0, stream>>>(whh0b, w_hh0, TH_, H_, TH_ * H_);
  cvt_pad_kernel<<<ew(TH_ * H_), 256, 0, stream>>>(wih1b, w_ih1, TH_, H_, TH_ * H_);
  cvt_pad_kernel<<<ew(TH_ * H_), 256, 0, stream>>>(whh1b, w_hh1, TH_, H_, TH_ * H_);
  cvt_pad_kernel<<<ew(NP_ * H_), 256, 0, stream>>>(web, w_energy, NOPS_, H_, NP_ * H_);
  cvt_pad_kernel<<<ew(512 * 512), 256, 0, stream>>>(wprojb, w_proj, IN_, 512, 512 * 512);
  cvt_pad_kernel<<<ew(OPE_PAD * 256), 256, 0, stream>>>(opeb, op_emb, NOPS_, 256, OPE_PAD * 256);
  cvt_pad_kernel<<<ew(H_ * 256), 256, 0, stream>>>(wopb, w_op, H_, 256, H_ * 256);
  pad_bias_kernel<<<ew(NP_), 256, 0, stream>>>(bep, b_energy, NOPS_, NP_);
  pad_bias_kernel<<<ew(512), 256, 0, stream>>>(bprojp, b_proj, IN_, 512);
  // ---- prep: state init
  concat_x0_kernel<<<ew(B_ * IN_), 256, 0, stream>>>(xb, geo, sem);
  zero_kernel<<<ew(B_ * H_), 256, 0, stream>>>((unsigned*)h0f, B_ * H_);
  zero_kernel<<<ew(B_ * H_), 256, 0, stream>>>((unsigned*)h1f, B_ * H_);
  zero_kernel<<<ew(B_ * H_ / 2), 256, 0, stream>>>((unsigned*)h0b, B_ * H_ / 2);
  zero_kernel<<<ew(B_ * H_ / 2), 256, 0, stream>>>((unsigned*)h1b, B_ * H_ / 2);
  zero_kernel<<<ew(NP_ * H_), 256, 0, stream>>>((unsigned*)oppf, NP_ * H_);
  // ---- prep: opp = op_embeds @ w_op^T  (M=1920 padded rows, N=256, K=256)
  gemm_bf16_kernel<<<dim3(1, OPE_PAD / 32), 128, 0, stream>>>(
      opeb, 256, wopb, 256, nullptr, oppf, nullptr, H_, 256, H_);
  oppstat_kernel<<<NP_, 256, 0, stream>>>(oppf, oppb, oppsq);

  // ---- 20 sequential steps
  for (int s = 0; s < STEPS_; ++s) {
    // GRU layer 0
    gemm_bf16_kernel<<<dim3(3, B_ / 32), 128, 0, stream>>>(
        xb, IN_, wih0b, IN_, b_ih0, GI, nullptr, TH_, IN_, TH_);
    gemm_bf16_kernel<<<dim3(3, B_ / 32), 128, 0, stream>>>(
        h0b, H_, whh0b, H_, b_hh0, GH, nullptr, TH_, H_, TH_);
    gates_kernel<<<B_, 256, 0, stream>>>(GI, GH, h0f, h0b, nullptr);
    // GRU layer 1
    gemm_bf16_kernel<<<dim3(3, B_ / 32), 128, 0, stream>>>(
        h0b, H_, wih1b, H_, b_ih1, GI, nullptr, TH_, H_, TH_);
    gemm_bf16_kernel<<<dim3(3, B_ / 32), 128, 0, stream>>>(
        h1b, H_, whh1b, H_, b_hh1, GH, nullptr, TH_, H_, TH_);
    gates_kernel<<<B_, 256, 0, stream>>>(GI, GH, h1f, h1b, hsq);
    // scores + argmax + next-x
    float* sc = scores_base + (size_t)s * B_ * NOPS_;
    scores_kernel<<<dim3(NP_ / 256, B_ / 32), 128, 0, stream>>>(
        h1b, web, oppb, bep, oppsq, hsq, sc);
    argmax_kernel<<<B_ / 8, 256, 0, stream>>>(
        sc, traj_base + (size_t)s * B_, h1b, op_emb, xcatb);
    gemm_bf16_kernel<<<dim3(2, B_ / 32), 128, 0, stream>>>(
        xcatb, 512, wprojb, 512, bprojp, nullptr, xb, IN_, 512, IN_);
  }
}
